// EdgeUpdateMLP_14336600834812
// MI455X (gfx1250) — matrix operand order repaired and checked by reading, not compiled, not run on hardware
//
#include <hip/hip_runtime.h>

#define N_EDGES  1600000
#define N_NODES  50000
#define NODE_DIM 64
#define EDGE_DIM 32
#define HIDDEN   128
#define OUT_DIM  64
#define IN_DIM   160   // EDGE_DIM + 2*NODE_DIM
#define HPAD     136   // padded bf16 row stride for h tile in LDS (bank-conflict-free)

// d_ws layout (bf16 elements):
//   [0, 20480)            W1 [128,160] row-major
//   [20480, 28672)        W2 [64,128]  row-major
//   [28672, 28672+3.2M)   node_features [50000,64] bf16
#define WS_W1   0
#define WS_W2   (HIDDEN * IN_DIM)
#define WS_NODE (HIDDEN * IN_DIM + OUT_DIM * HIDDEN)
#define N_NODE_ELEMS (N_NODES * NODE_DIM)

typedef __attribute__((ext_vector_type(16))) __bf16       v16bf;
typedef __attribute__((ext_vector_type(2)))  __bf16       v2bf;
typedef __attribute__((ext_vector_type(8)))  float        v8f;
typedef __attribute__((ext_vector_type(4)))  float        v4f;
typedef __attribute__((ext_vector_type(2)))  float        v2f;
typedef __attribute__((ext_vector_type(4)))  unsigned int v4u;

union Frag {
  v16bf v;
  unsigned short u[16];
  v4u q[2];
};

// Hardware bf16 convert (RNE): lowers to v_cvt_pk_bf16_f32-class ops.
__device__ __forceinline__ unsigned short f2bf(float f) {
  union { __bf16 b; unsigned short s; } cv;
  cv.b = (__bf16)f;
  return cv.s;
}

// Packed pair convert: one v_cvt_pk_bf16_f32 for two floats.
__device__ __forceinline__ unsigned int f2bf_pk(float lo, float hi) {
  v2f f; f.x = lo; f.y = hi;
  v2bf b = __builtin_convertvector(f, v2bf);
  union { v2bf b; unsigned int u; } cv; cv.b = b;
  return cv.u;
}

__device__ __forceinline__ void cvt8(unsigned short* d, const float* p) {
  v4f a = *(const v4f*)p;
  v4f b = *(const v4f*)(p + 4);
  d[0] = f2bf(a.x); d[1] = f2bf(a.y); d[2] = f2bf(a.z); d[3] = f2bf(a.w);
  d[4] = f2bf(b.x); d[5] = f2bf(b.y); d[6] = f2bf(b.z); d[7] = f2bf(b.w);
}

// One-time fp32 -> bf16 conversion of W1, W2 and the node-feature table into
// d_ws (node table is reused ~64x and stays L2-resident).
__global__ void convert_kernel(const float* __restrict__ W1,
                               const float* __restrict__ W2,
                               const float* __restrict__ nodef,
                               unsigned short* __restrict__ ws) {
  int i = blockIdx.x * 256 + threadIdx.x;
  if (i < HIDDEN * IN_DIM)  ws[WS_W1 + i] = f2bf(W1[i]);
  if (i < OUT_DIM * HIDDEN) ws[WS_W2 + i] = f2bf(W2[i]);
  if (i < N_NODE_ELEMS)     ws[WS_NODE + i] = f2bf(nodef[i]);
}

// 128 threads = 4 waves; each wave owns TWO 16-edge M-tiles (32 edges) so every
// B (weight) fragment load feeds two independent WMMAs.
__global__ void __launch_bounds__(128)
edge_mlp_kernel(const int*   __restrict__ eidx,
                const float* __restrict__ edgef,
                const float* __restrict__ b1,
                const float* __restrict__ b2,
                const unsigned short* __restrict__ wsbf,
                float* __restrict__ out) {
  __shared__ __attribute__((aligned(32))) unsigned short hbuf[4 * 32 * HPAD];

  const int lane = threadIdx.x & 31;
  const int wave = threadIdx.x >> 5;      // wave32, 0..3
  const int m    = lane & 15;             // tile row (A) / tile column (B,C,D)
  const int hL   = lane >> 4;             // half-select per WMMA layout
  const int e0   = (blockIdx.x * 4 + wave) * 32;
  if (e0 + 32 > N_EDGES) return;          // E is a multiple of 128; defensive only

  // Per-lane gather pointers for the two edge rows this lane covers.
  const int ea = e0 + m;
  const int eb = e0 + 16 + m;
  const int sia = eidx[ea],           tia = eidx[N_EDGES + ea];
  const int sib = eidx[eb],           tib = eidx[N_EDGES + eb];
  const float*          ef_a = edgef + (size_t)ea * EDGE_DIM;
  const float*          ef_b = edgef + (size_t)eb * EDGE_DIM;
  const unsigned short* nbf  = wsbf + WS_NODE;
  const unsigned short* s_a  = nbf + (size_t)sia * NODE_DIM;
  const unsigned short* t_a  = nbf + (size_t)tia * NODE_DIM;
  const unsigned short* s_b  = nbf + (size_t)sib * NODE_DIM;
  const unsigned short* t_b  = nbf + (size_t)tib * NODE_DIM;

  const unsigned short* w1bf = wsbf + WS_W1;
  const unsigned short* w2bf = wsbf + WS_W2;

  // ---------------- layer 1: 2x [16 x 160] @ [160 x 128] ----------------
  v8f acc1[2][8] = {};
  #pragma unroll
  for (int kb = 0; kb < 5; ++kb) {
    Frag A0, A1;  // elems 0..7 = X[m][kb*32+hL*8+j]; 8..15 = X[m][kb*32+16+hL*8+j]
    if (kb == 0) {
      // edge features: fp32, used once each -> convert in flight (hw pk-cvt)
      cvt8(A0.u,     ef_a + hL * 8);
      cvt8(A0.u + 8, ef_a + 16 + hL * 8);
      cvt8(A1.u,     ef_b + hL * 8);
      cvt8(A1.u + 8, ef_b + 16 + hL * 8);
    } else {
      // node features: already bf16 -> raw 16B loads (L2-resident gathers)
      const int off = ((kb < 3) ? (kb - 1) : (kb - 3)) * 32;
      const unsigned short* ba = (kb < 3) ? s_a : t_a;
      const unsigned short* bb = (kb < 3) ? s_b : t_b;
      A0.q[0] = *(const v4u*)(ba + off + hL * 8);
      A0.q[1] = *(const v4u*)(ba + off + 16 + hL * 8);
      A1.q[0] = *(const v4u*)(bb + off + hL * 8);
      A1.q[1] = *(const v4u*)(bb + off + 16 + hL * 8);
    }
    #pragma unroll
    for (int nt = 0; nt < 8; ++nt) {
      Frag B;  // elems 0..7 = W1[nt*16+m][kb*32+hL*8+j]; 8..15 = W1[nt*16+m][kb*32+16+hL*8+j]
      const unsigned short* bp =
          w1bf + (size_t)(nt * 16 + m) * IN_DIM + kb * 32;
      B.q[0] = *(const v4u*)(bp + hL * 8);
      B.q[1] = *(const v4u*)(bp + 16 + hL * 8);
      // One B fragment feeds two independent WMMAs (no D->A/B hazard between them).
      acc1[0][nt] = __builtin_amdgcn_wmma_f32_16x16x32_bf16(
          false, A0.v, false, B.v, (short)0, acc1[0][nt], false, false);
      acc1[1][nt] = __builtin_amdgcn_wmma_f32_16x16x32_bf16(
          false, A1.v, false, B.v, (short)0, acc1[1][nt], false, false);
    }
  }

  // bias + relu, stage 32x128 h tile (bf16) in this wave's private LDS slice.
  // Rows r and r+1 (same column) are converted with one packed cvt; the high
  // half goes out via the d16_hi 16-bit store pattern.
  unsigned short* myh = hbuf + wave * (32 * HPAD);
  #pragma unroll
  for (int nt = 0; nt < 8; ++nt) {
    const float bias = b1[nt * 16 + m];             // D column = m
    #pragma unroll
    for (int t = 0; t < 2; ++t) {
      unsigned short* base = myh + (t * 16 + 8 * hL) * HPAD + nt * 16 + m;
      #pragma unroll
      for (int r = 0; r < 8; r += 2) {              // D rows r,r+1 (+8*hL)
        float v0 = acc1[t][nt][r]     + bias;
        float v1 = acc1[t][nt][r + 1] + bias;
        v0 = v0 > 0.0f ? v0 : 0.0f;
        v1 = v1 > 0.0f ? v1 : 0.0f;
        const unsigned int u = f2bf_pk(v0, v1);     // one v_cvt_pk_bf16_f32
        base[(size_t)r * HPAD]       = (unsigned short)u;
        base[(size_t)(r + 1) * HPAD] = (unsigned short)(u >> 16);
      }
    }
  }
  // Same-wave DS ops are architecturally in-order; no barrier needed.

  // ---------------- layer 2: 2x [16 x 128] @ [128 x 64] ----------------
  v8f acc2[2][4] = {};
  const unsigned short* hrow0 = myh + m * HPAD;          // tile 0, A row = m
  const unsigned short* hrow1 = myh + (16 + m) * HPAD;   // tile 1
  #pragma unroll
  for (int kb = 0; kb < 4; ++kb) {
    Frag A0, A1;                                    // raw 16B LDS loads, already bf16
    A0.q[0] = *(const v4u*)(hrow0 + kb * 32      + hL * 8);
    A0.q[1] = *(const v4u*)(hrow0 + kb * 32 + 16 + hL * 8);
    A1.q[0] = *(const v4u*)(hrow1 + kb * 32      + hL * 8);
    A1.q[1] = *(const v4u*)(hrow1 + kb * 32 + 16 + hL * 8);
    #pragma unroll
    for (int nt = 0; nt < 4; ++nt) {
      Frag B;  // elems 0..7 = W2[nt*16+m][kb*32+hL*8+j]; 8..15 = W2[nt*16+m][kb*32+16+hL*8+j]
      const unsigned short* bp =
          w2bf + (size_t)(nt * 16 + m) * HIDDEN + kb * 32;
      B.q[0] = *(const v4u*)(bp + hL * 8);
      B.q[1] = *(const v4u*)(bp + 16 + hL * 8);
      acc2[0][nt] = __builtin_amdgcn_wmma_f32_16x16x32_bf16(
          false, A0.v, false, B.v, (short)0, acc2[0][nt], false, false);
      acc2[1][nt] = __builtin_amdgcn_wmma_f32_16x16x32_bf16(
          false, A1.v, false, B.v, (short)0, acc2[1][nt], false, false);
    }
  }

  // bias + store out [E, 64] fp32 (lanes 0-15 / 16-31 each cover 64B row-segments)
  #pragma unroll
  for (int nt = 0; nt < 4; ++nt) {
    const float bias = b2[nt * 16 + m];
    #pragma unroll
    for (int t = 0; t < 2; ++t) {
      #pragma unroll
      for (int r = 0; r < 8; ++r) {
        out[(size_t)(e0 + t * 16 + r + 8 * hL) * OUT_DIM + nt * 16 + m] =
            acc2[t][nt][r] + bias;
      }
    }
  }
}

extern "C" void kernel_launch(void* const* d_in, const int* in_sizes, int n_in,
                              void* d_out, int out_size, void* d_ws, size_t ws_size,
                              hipStream_t stream) {
  const int*   eidx  = (const int*)  d_in[0];   // edge_index [2, E] int32
  const float* nodef = (const float*)d_in[1];   // node_features [N, 64]
  const float* edgef = (const float*)d_in[2];   // edge_features [E, 32]
  const float* W1    = (const float*)d_in[3];   // [128, 160]
  const float* b1    = (const float*)d_in[4];   // [128]
  const float* W2    = (const float*)d_in[5];   // [64, 128]
  const float* b2    = (const float*)d_in[6];   // [64]
  unsigned short* ws = (unsigned short*)d_ws;   // ~6.2 MB bf16 weights + node table
  float* out = (float*)d_out;                   // [E, 64]

  (void)in_sizes; (void)n_in; (void)out_size; (void)ws_size;

  // Covers node table (3.2M elems) and, via guards, both weight matrices.
  convert_kernel<<<(N_NODE_ELEMS + 255) / 256, 256, 0, stream>>>(W1, W2, nodef, ws);

  const int blocks = (N_EDGES + 127) / 128;     // 128 edges per 4-wave block
  edge_mlp_kernel<<<blocks, 128, 0, stream>>>(eidx, edgef, b1, b2, ws, out);
}
